// MultiHeadAttention_81905026335102
// MI455X (gfx1250) — matrix operand order repaired and checked by reading, not compiled, not run on hardware
//
#include <hip/hip_runtime.h>
#include <hip/hip_bf16.h>

#define Bq 4
#define Sq 2048
#define Eq 128
#define Hq 8
#define HEq (Hq * Eq)                     // 1024
#define BSEq (Bq * Sq * Eq)               // 1048576
#define BHSEq (Bq * Hq * Sq * Eq)         // 8388608
#define WSZq (Hq * Eq * Eq)               // 131072

typedef __attribute__((ext_vector_type(16))) __bf16 v16bf;
typedef __attribute__((ext_vector_type(8)))  float  v8f;

#define DEVI __device__ __forceinline__

union ABFrag { v16bf v; uint4 q[2]; };

// Fragment from two 8-element (16B) chunks (A-layout: split halves)
static DEVI v16bf frag2(const __bf16* p0, const __bf16* p1) {
    ABFrag f;
    f.q[0] = *(const uint4*)p0;
    f.q[1] = *(const uint4*)p1;
    return f.v;
}
// Fragment from one contiguous 16-element (32B) chunk (unused)
static DEVI v16bf frag1(const __bf16* p) { return frag2(p, p + 8); }

static DEVI __bf16 f2bf(float f) {
    __hip_bfloat16 h = __float2bfloat16(f);
    __bf16 r;
    __builtin_memcpy(&r, &h, 2);
    return r;
}

static DEVI v8f wmma_bf16(v16bf a, v16bf b, v8f c) {
    return __builtin_amdgcn_wmma_f32_16x16x32_bf16(
        /*neg_a=*/false, a, /*neg_b=*/false, b,
        /*c_mod=*/(short)0, c, /*reuse_a=*/false, /*reuse_b=*/false);
}

// ---------------------------------------------------------------------------
// Kernel 0: convert q -> bf16; transpose+convert Wq/Wk/Wv -> [h][out][in] bf16;
//           transpose+convert Wo -> [out=128][in=1024] bf16.
// ---------------------------------------------------------------------------
__global__ void prep_kernel(const float* __restrict__ q,
                            const float* __restrict__ Wq,
                            const float* __restrict__ Wk,
                            const float* __restrict__ Wv,
                            const float* __restrict__ Wo,
                            __bf16* __restrict__ qb,
                            __bf16* __restrict__ WT,   // 3 * H*E*E, [mat][h][out][in]
                            __bf16* __restrict__ WoT)  // [128][1024]
{
    const int tid    = blockIdx.x * blockDim.x + threadIdx.x;
    const int stride = gridDim.x * blockDim.x;

    for (int i = tid; i < BSEq; i += stride) qb[i] = f2bf(q[i]);

    for (int i = tid; i < WSZq; i += stride) {
        int h = i / (Eq * Eq);
        int r = i - h * (Eq * Eq);
        int k = r / Eq;            // input dim
        int n = r - k * Eq;        // output dim
        int d = h * (Eq * Eq) + n * Eq + k;
        WT[d]            = f2bf(Wq[i]);
        WT[WSZq + d]     = f2bf(Wk[i]);
        WT[2 * WSZq + d] = f2bf(Wv[i]);
    }

    for (int i = tid; i < HEq * Eq; i += stride) {   // Wo is [1024][128]
        int k = i / Eq;
        int n = i - k * Eq;
        WoT[n * HEq + k] = f2bf(Wo[i]);
    }
}

// ---------------------------------------------------------------------------
// Kernel 1: QKV projections. One 16x16 output tile per wave, K=128 (4 WMMAs).
// Output bf16 [mat][b][h][s][e].
// ---------------------------------------------------------------------------
__global__ void qkv_proj_kernel(const __bf16* __restrict__ qb,
                                const __bf16* __restrict__ WT,
                                const float* __restrict__ bq,
                                const float* __restrict__ bk,
                                const float* __restrict__ bv,
                                __bf16* __restrict__ QKV)
{
    int wid  = (blockIdx.x * blockDim.x + threadIdx.x) >> 5;
    const int lane = threadIdx.x & 31;
    const int ln   = lane & 15;
    const int half = lane >> 4;

    const int ct  = wid & 7;   wid >>= 3;   // column tile (E/16)
    const int rt  = wid & 127; wid >>= 7;   // row tile (S/16)
    const int bh  = wid & 31;  wid >>= 5;   // b*H + h
    const int mat = wid;                    // 0=Q 1=K 2=V
    const int b   = bh >> 3;
    const int h   = bh & 7;

    const float* bias = (mat == 0) ? bq : (mat == 1) ? bk : bv;
    const float bval  = bias[h * Eq + ct * 16 + ln];

    v8f c;
#pragma unroll
    for (int i = 0; i < 8; ++i) c[i] = bval;

    const __bf16* arow = qb + (size_t)(b * Sq + rt * 16 + ln) * Eq;
    const __bf16* wrow = WT + (size_t)mat * WSZq + (size_t)h * (Eq * Eq)
                            + (size_t)(ct * 16 + ln) * Eq;
#pragma unroll
    for (int kc = 0; kc < 4; ++kc) {
        v16bf a = frag2(arow + kc * 32 + half * 8,
                        arow + kc * 32 + 16 + half * 8);
        v16bf w = frag2(wrow + kc * 32 + half * 8, wrow + kc * 32 + 16 + half * 8);
        c = wmma_bf16(a, w, c);
    }

    __bf16* out = QKV + (size_t)mat * BHSEq + (size_t)bh * Sq * Eq;
#pragma unroll
    for (int r = 0; r < 8; ++r) {
        int m = half * 8 + r;
        out[(size_t)(rt * 16 + m) * Eq + ct * 16 + ln] = f2bf(c[r]);
    }
}

// ---------------------------------------------------------------------------
// Kernel 2: flash attention, 64-token K/V tiles.
// Block = (b,h, 128 Q rows); wave = 16 Q rows, Q resident in VGPRs.
// Per tile: 16 WMMAs scores + 16 WMMAs P@V, one softmax pass, 2 barriers.
// Writes Z bf16 [b][s][h*E + e] (head-concat layout for the out-projection).
// ---------------------------------------------------------------------------
__global__ void attn_kernel(const __bf16* __restrict__ QKV,
                            __bf16* __restrict__ Z)
{
    const __bf16* Qd = QKV;
    const __bf16* Kd = QKV + (size_t)BHSEq;
    const __bf16* Vd = QKV + (size_t)2 * BHSEq;

    const int qblk = blockIdx.x & 15;
    const int bh   = blockIdx.x >> 4;
    const int b    = bh >> 3;
    const int h    = bh & 7;

    const int w    = threadIdx.x >> 5;
    const int lane = threadIdx.x & 31;
    const int ln   = lane & 15;
    const int half = lane >> 4;
    const int qr0  = qblk * 128 + w * 16;

    __shared__ __align__(16) __bf16 Ks[64 * 128];      // [token][feat]       16KB
    __shared__ __align__(16) __bf16 Vt[128 * 64];      // [feat][token]       16KB
    __shared__ __align__(16) __bf16 Ps[8][16 * 64];    // per-wave P [m][k]   16KB

    // Resident Q fragments (A layout), 4 chunks of K=32 over E=128
    const __bf16* qrow = Qd + (size_t)(bh * Sq + qr0 + ln) * Eq;
    v16bf qf[4];
#pragma unroll
    for (int kc = 0; kc < 4; ++kc)
        qf[kc] = frag2(qrow + kc * 32 + half * 8, qrow + kc * 32 + 16 + half * 8);

    v8f Oacc[8];
#pragma unroll
    for (int nf = 0; nf < 8; ++nf)
#pragma unroll
        for (int i = 0; i < 8; ++i) Oacc[nf][i] = 0.0f;

    float mi[8], li[8];
#pragma unroll
    for (int r = 0; r < 8; ++r) { mi[r] = -1e30f; li[r] = 0.0f; }

    const float scale = 0.088388347648318447f;  // 1/sqrt(128)

    const int tok = threadIdx.x >> 2;   // 0..63  (staging decomposition)
    const int cc  = threadIdx.x & 3;    // 0..3   (32-feature chunk)

    for (int j = 0; j < Sq / 64; ++j) {
        __syncthreads();
        {   // cooperative staging: 64 tokens x 128 feats, 32 bf16 per thread
            const __bf16* ksrc = Kd + (size_t)(bh * Sq + j * 64 + tok) * Eq + cc * 32;
#pragma unroll
            for (int u = 0; u < 4; ++u)
                *(uint4*)&Ks[tok * 128 + cc * 32 + u * 8] = *(const uint4*)(ksrc + u * 8);
            const __bf16* vsrc = Vd + (size_t)(bh * Sq + j * 64 + tok) * Eq + cc * 32;
#pragma unroll
            for (int i = 0; i < 32; ++i)
                Vt[(cc * 32 + i) * 64 + tok] = vsrc[i];
        }
        __syncthreads();

        // prefetch next tile while this one computes
        if (j + 1 < Sq / 64) {
            __builtin_prefetch(Kd + (size_t)(bh * Sq + (j + 1) * 64 + tok) * Eq + cc * 32, 0, 0);
            __builtin_prefetch(Vd + (size_t)(bh * Sq + (j + 1) * 64 + tok) * Eq + cc * 32, 0, 0);
        }

        // S = Q @ K^T for 4 N-subtiles of the 64-token tile (16 WMMAs)
        v8f s[4];
#pragma unroll
        for (int nh = 0; nh < 4; ++nh)
#pragma unroll
            for (int i = 0; i < 8; ++i) s[nh][i] = 0.0f;
#pragma unroll
        for (int kc = 0; kc < 4; ++kc)
#pragma unroll
            for (int nh = 0; nh < 4; ++nh)
                s[nh] = wmma_bf16(qf[kc],
                                  frag2(&Ks[(nh * 16 + ln) * 128 + kc * 32 + half * 8], &Ks[(nh * 16 + ln) * 128 + kc * 32 + 16 + half * 8]),
                                  s[nh]);

        // online softmax: scale, row max/sum via 16-lane shuffle reductions,
        // exp in place (s becomes P)
#pragma unroll
        for (int nh = 0; nh < 4; ++nh)
#pragma unroll
            for (int i = 0; i < 8; ++i) s[nh][i] *= scale;

#pragma unroll
        for (int r = 0; r < 8; ++r) {
            float tmax = fmaxf(fmaxf(s[0][r], s[1][r]), fmaxf(s[2][r], s[3][r]));
            tmax = fmaxf(tmax, __shfl_xor(tmax, 1, 32));
            tmax = fmaxf(tmax, __shfl_xor(tmax, 2, 32));
            tmax = fmaxf(tmax, __shfl_xor(tmax, 4, 32));
            tmax = fmaxf(tmax, __shfl_xor(tmax, 8, 32));
            float mnew  = fmaxf(mi[r], tmax);
            float alpha = __expf(mi[r] - mnew);
#pragma unroll
            for (int nh = 0; nh < 4; ++nh) s[nh][r] = __expf(s[nh][r] - mnew);
            float rs = (s[0][r] + s[1][r]) + (s[2][r] + s[3][r]);
            rs += __shfl_xor(rs, 1, 32);
            rs += __shfl_xor(rs, 2, 32);
            rs += __shfl_xor(rs, 4, 32);
            rs += __shfl_xor(rs, 8, 32);
            li[r] = li[r] * alpha + rs;
            mi[r] = mnew;
#pragma unroll
            for (int nf = 0; nf < 8; ++nf) Oacc[nf][r] *= alpha;
        }

        // P: C-layout -> A-layout via per-wave LDS bounce
#pragma unroll
        for (int r = 0; r < 8; ++r) {
            int m = half * 8 + r;
#pragma unroll
            for (int nh = 0; nh < 4; ++nh)
                Ps[w][m * 64 + nh * 16 + ln] = f2bf(s[nh][r]);
        }
        v16bf pf0 = frag2(&Ps[w][ln * 64 + half * 8],
                          &Ps[w][ln * 64 + 16 + half * 8]);
        v16bf pf1 = frag2(&Ps[w][ln * 64 + 32 + half * 8],
                          &Ps[w][ln * 64 + 48 + half * 8]);

        // O += P @ V  (V transposed in LDS -> contiguous B fragments, 16 WMMAs)
#pragma unroll
        for (int nf = 0; nf < 8; ++nf) {
            Oacc[nf] = wmma_bf16(pf0, frag2(&Vt[(nf * 16 + ln) * 64 + half * 8], &Vt[(nf * 16 + ln) * 64 + 16 + half * 8]),
                                 Oacc[nf]);
            Oacc[nf] = wmma_bf16(pf1, frag2(&Vt[(nf * 16 + ln) * 64 + 32 + half * 8], &Vt[(nf * 16 + ln) * 64 + 48 + half * 8]),
                                 Oacc[nf]);
        }
    }

    // normalize and write Z (head-concat layout)
    float inv[8];
#pragma unroll
    for (int r = 0; r < 8; ++r) inv[r] = 1.0f / li[r];
#pragma unroll
    for (int nf = 0; nf < 8; ++nf)
#pragma unroll
        for (int r = 0; r < 8; ++r) {
            int m = half * 8 + r;
            Z[(size_t)(b * Sq + qr0 + m) * HEq + h * Eq + nf * 16 + ln] =
                f2bf(Oacc[nf][r] * inv[r]);
        }
}

// ---------------------------------------------------------------------------
// Kernel 3: output projection. Y[8192,128] = Z[8192,1024] @ WoT^T + bo (fp32 out)
// ---------------------------------------------------------------------------
__global__ void out_proj_kernel(const __bf16* __restrict__ Z,
                                const __bf16* __restrict__ WoT,
                                const float* __restrict__ bo,
                                float* __restrict__ out)
{
    int wid  = (blockIdx.x * blockDim.x + threadIdx.x) >> 5;
    const int lane = threadIdx.x & 31;
    const int ln   = lane & 15;
    const int half = lane >> 4;

    const int ct = wid & 7;    // N tile (E/16)
    const int rt = wid >> 3;   // M tile (B*S/16)

    const float bval = bo[ct * 16 + ln];
    v8f c;
#pragma unroll
    for (int i = 0; i < 8; ++i) c[i] = bval;

    const __bf16* arow = Z   + (size_t)(rt * 16 + ln) * HEq;
    const __bf16* wrow = WoT + (size_t)(ct * 16 + ln) * HEq;
#pragma unroll
    for (int kc = 0; kc < 32; ++kc) {
        v16bf a = frag2(arow + kc * 32 + half * 8,
                        arow + kc * 32 + 16 + half * 8);
        v16bf w = frag2(wrow + kc * 32 + half * 8, wrow + kc * 32 + 16 + half * 8);
        c = wmma_bf16(a, w, c);
    }
#pragma unroll
    for (int r = 0; r < 8; ++r) {
        int m = half * 8 + r;
        out[(size_t)(rt * 16 + m) * Eq + ct * 16 + ln] = c[r];
    }
}

// ---------------------------------------------------------------------------
extern "C" void kernel_launch(void* const* d_in, const int* in_sizes, int n_in,
                              void* d_out, int out_size, void* d_ws, size_t ws_size,
                              hipStream_t stream) {
    const float* q  = (const float*)d_in[0];
    const float* Wq = (const float*)d_in[1];
    const float* bq = (const float*)d_in[2];
    const float* Wk = (const float*)d_in[3];
    const float* bk = (const float*)d_in[4];
    const float* Wv = (const float*)d_in[5];
    const float* bv = (const float*)d_in[6];
    const float* Wo = (const float*)d_in[7];
    const float* bo = (const float*)d_in[8];
    float* out = (float*)d_out;

    char* ws = (char*)d_ws;
    size_t off = 0;
    __bf16* qb  = (__bf16*)(ws + off); off += (size_t)BSEq * 2;
    __bf16* WT  = (__bf16*)(ws + off); off += (size_t)3 * WSZq * 2;
    __bf16* WoT = (__bf16*)(ws + off); off += (size_t)WSZq * 2;
    __bf16* QKV = (__bf16*)(ws + off); off += (size_t)3 * BHSEq * 2;
    __bf16* Zb  = (__bf16*)(ws + off); off += (size_t)BHSEq * 2;

    prep_kernel<<<2048, 256, 0, stream>>>(q, Wq, Wk, Wv, Wo, qb, WT, WoT);

    // tiles = 3 mats * 32 bh * 128 row-tiles * 8 col-tiles = 98304 waves
    qkv_proj_kernel<<<98304 / 8, 256, 0, stream>>>(qb, WT, bq, bk, bv, QKV);

    // 32 (b,h) * 16 q-blocks of 128 rows
    attn_kernel<<<512, 256, 0, stream>>>(QKV, Zb);

    // tiles = 512 row-tiles * 8 col-tiles = 4096 waves
    out_proj_kernel<<<4096 / 8, 256, 0, stream>>>(Zb, WoT, bo, out);
}
